// TrueSparseAttention_13932873908462
// MI455X (gfx1250) — matrix operand order repaired and checked by reading, not compiled, not run on hardware
//
#include <hip/hip_runtime.h>
#include <hip/hip_bf16.h>
#include <stdint.h>

#define B_    2
#define S_    2048
#define D_    2048
#define H_    16
#define KVH_  8
#define HD_   128
#define TOPK_ 1024

typedef __attribute__((ext_vector_type(16))) __bf16 v16bf;
typedef __attribute__((ext_vector_type(8)))  float  v8f;

union FragBF { uint4 q[2]; v16bf v; };

__device__ __forceinline__ unsigned short f2bf(float f) {
  unsigned u = __builtin_bit_cast(unsigned, f);
  unsigned r = u + 0x7FFFu + ((u >> 16) & 1u);
  return (unsigned short)(r >> 16);
}
// monotonic orderable key for fp32 (larger float -> larger key)
__device__ __forceinline__ unsigned fkey(float f) {
  unsigned u = __builtin_bit_cast(unsigned, f);
  return (u & 0x80000000u) ? ~u : (u | 0x80000000u);
}
__device__ __forceinline__ float fkey_inv(unsigned k) {
  unsigned u = (k & 0x80000000u) ? (k ^ 0x80000000u) : ~k;
  return __builtin_bit_cast(float, u);
}
__device__ __forceinline__ int wred_sum_i(int v) {
  #pragma unroll
  for (int o = 16; o > 0; o >>= 1) v += __shfl_xor(v, o, 32);
  return v;
}
__device__ __forceinline__ float wred_sum_f(float v) {
  #pragma unroll
  for (int o = 16; o > 0; o >>= 1) v += __shfl_xor(v, o, 32);
  return v;
}
__device__ __forceinline__ unsigned wred_max_u(unsigned v) {
  #pragma unroll
  for (int o = 16; o > 0; o >>= 1) {
    unsigned t = (unsigned)__shfl_xor((int)v, o, 32);
    v = (t > v) ? t : v;
  }
  return v;
}

// Async global->LDS copy of 32 contiguous bytes per lane (2 x b128, ASYNCcnt)
__device__ __forceinline__ void async_copy32(unsigned lds_byte_off,
                                             const unsigned short* gptr) {
  asm volatile(
      "global_load_async_to_lds_b128 %0, %1, off\n\t"
      "global_load_async_to_lds_b128 %0, %1, off offset:16"
      :: "v"(lds_byte_off), "v"(gptr) : "memory");
}
__device__ __forceinline__ void wait_async_le4() {
  asm volatile("s_wait_asynccnt 0x4" ::: "memory");
}
__device__ __forceinline__ void wait_async_0() {
  asm volatile("s_wait_asynccnt 0x0" ::: "memory");
}

// ---------------------------------------------------------------------------
// Elementwise fp32 -> bf16 (4 elements / thread). n must be multiple of 1024.
// ---------------------------------------------------------------------------
__global__ __launch_bounds__(256) void to_bf16(const float* __restrict__ src,
                                               unsigned short* __restrict__ dst) {
  size_t i = ((size_t)blockIdx.x * 256 + threadIdx.x) * 4;
  float4 f = *(const float4*)(src + i);
  ushort4 o;
  o.x = f2bf(f.x); o.y = f2bf(f.y); o.z = f2bf(f.z); o.w = f2bf(f.w);
  *(ushort4*)(dst + i) = o;
}

// ---------------------------------------------------------------------------
// GEMM: C[M,N] = A[M,K] @ W[N,K]^T + bias[N].  A, W bf16; C fp32.
// Block tile 128x128, K-step 32. Double-buffered LDS staging via
// GLOBAL_LOAD_ASYNC_TO_LDS_B128 (ASYNCcnt), 8 waves x (2x4) WMMA tiles.
// ---------------------------------------------------------------------------
__global__ __launch_bounds__(256) void gemm_bias(
    const unsigned short* __restrict__ A, const unsigned short* __restrict__ W,
    const float* __restrict__ bias, float* __restrict__ C,
    int M, int N, int K) {
  __shared__ unsigned short At[2][128 * 32];
  __shared__ unsigned short Wt[2][128 * 32];
  const int tid  = threadIdx.x;
  const int lane = tid & 31;
  const int wave = tid >> 5;
  const int hi   = lane >> 4;
  const int l15  = lane & 15;
  const int m0   = blockIdx.y * 128;
  const int n0   = blockIdx.x * 128;
  const int wm   = wave >> 1;   // 0..3 (M)
  const int wn   = wave & 1;    // 0..1 (N)
  const int row  = tid >> 1;    // 0..127 staging row
  const int half = tid & 1;     // 16-element half

  const v8f z8 = {0.f, 0.f, 0.f, 0.f, 0.f, 0.f, 0.f, 0.f};
  v8f acc[2][4];
  #pragma unroll
  for (int mi = 0; mi < 2; ++mi)
    #pragma unroll
    for (int ni = 0; ni < 4; ++ni) acc[mi][ni] = z8;

  const int stg = row * 32 + half * 16;                  // bf16 elements
  const unsigned short* gA0 = A + (size_t)(m0 + row) * K + half * 16;
  const unsigned short* gW0 = W + (size_t)(n0 + row) * K + half * 16;

  // prologue: stage tile 0 into buffer 0
  async_copy32((unsigned)(size_t)&At[0][stg], gA0);
  async_copy32((unsigned)(size_t)&Wt[0][stg], gW0);

  const int nk = K >> 5;
  for (int t = 0; t < nk; ++t) {
    const int buf = t & 1;
    if (t + 1 < nk) {
      async_copy32((unsigned)(size_t)&At[buf ^ 1][stg], gA0 + (t + 1) * 32);
      async_copy32((unsigned)(size_t)&Wt[buf ^ 1][stg], gW0 + (t + 1) * 32);
      wait_async_le4();   // tile t has landed; tile t+1 still in flight
    } else {
      wait_async_0();
    }
    __syncthreads();

    FragBF a[2], b[4];
    #pragma unroll
    for (int mi = 0; mi < 2; ++mi) {
      int mr = wm * 32 + mi * 16 + l15;
      a[mi].q[0] = *(const uint4*)&At[buf][mr * 32 + hi * 8];
      a[mi].q[1] = *(const uint4*)&At[buf][mr * 32 + 16 + hi * 8];
    }
    #pragma unroll
    for (int ni = 0; ni < 4; ++ni) {
      int nr = wn * 64 + ni * 16 + l15;
      b[ni].q[0] = *(const uint4*)&Wt[buf][nr * 32 + hi * 8];
      b[ni].q[1] = *(const uint4*)&Wt[buf][nr * 32 + 16 + hi * 8];
    }
    #pragma unroll
    for (int mi = 0; mi < 2; ++mi)
      #pragma unroll
      for (int ni = 0; ni < 4; ++ni)
        acc[mi][ni] = __builtin_amdgcn_wmma_f32_16x16x32_bf16(
            false, a[mi].v, false, b[ni].v, (short)0, acc[mi][ni], false, false);
    __syncthreads();   // all waves done reading buf before it is re-staged
  }

  #pragma unroll
  for (int mi = 0; mi < 2; ++mi)
    #pragma unroll
    for (int ni = 0; ni < 4; ++ni) {
      int gm = m0 + wm * 32 + mi * 16 + hi * 8;
      int gn = n0 + wn * 64 + ni * 16 + l15;
      float bv = bias[gn];
      #pragma unroll
      for (int r = 0; r < 8; ++r)
        C[(size_t)(gm + r) * N + gn] = acc[mi][ni][r] + bv;
    }
}

// ---------------------------------------------------------------------------
// RoPE + layout pack: Q->(B,H,S,HD) bf16, K->(B,KVH,S,HD) bf16 (both roped),
// V -> transposed (B,KVH,HD,S) bf16 for contiguous B-fragments in PV.
// hs slot: 0..15 = Q head, 16..23 = K head, 24..31 = V head.
// ---------------------------------------------------------------------------
__global__ __launch_bounds__(256) void rope_pack(
    const float* __restrict__ qf, const float* __restrict__ kf,
    const float* __restrict__ vf, const int* __restrict__ pos_ids,
    unsigned short* __restrict__ qb16, unsigned short* __restrict__ kb16,
    unsigned short* __restrict__ vt16) {
  size_t idx = (size_t)blockIdx.x * 256 + threadIdx.x;  // 2^24 total
  int d  = (int)(idx & (HD_ - 1));
  int hs = (int)((idx >> 7) & 31);
  int s  = (int)((idx >> 12) & (S_ - 1));
  int b  = (int)(idx >> 23);
  float p = (float)pos_ids[b * S_ + s];

  if (hs < 24) {
    const float* src;
    if (hs < 16) src = qf + ((size_t)(b * S_ + s)) * D_ + hs * HD_;
    else         src = kf + ((size_t)(b * S_ + s)) * (KVH_ * HD_) + (hs - 16) * HD_;
    int i = d & 63;
    float ang = p * powf(1000000.0f, -(float)i * (1.0f / 64.0f));
    float sn, c;
    sincosf(ang, &sn, &c);
    float x = src[d];
    float other = (d < 64) ? -src[d + 64] : src[d - 64];
    unsigned short bv = f2bf(x * c + other * sn);
    if (hs < 16)
      qb16[(((size_t)(b * H_ + hs)) * S_ + s) * HD_ + d] = bv;
    else
      kb16[(((size_t)(b * KVH_ + (hs - 16))) * S_ + s) * HD_ + d] = bv;
  } else {
    int kv = hs - 24;
    float x = vf[((size_t)(b * S_ + s)) * (KVH_ * HD_) + kv * HD_ + d];
    vt16[(((size_t)(b * KVH_ + kv)) * HD_ + d) * S_ + s] = f2bf(x);
  }
}

// ---------------------------------------------------------------------------
// Attention core: one workgroup per (b, h, 16-query tile).
// Phase 1: scores (16 x 2048) via WMMA -> LDS fp32 (128 KB dynamic LDS).
// Phase 2: exact 1024-th-largest per row (bitwise binary search on monotone
//          keys, wave32 reductions), softmax with ties kept, bf16 P -> LDS.
// Phase 3: out = P @ V via WMMA (V pre-transposed), bf16 to (B,S,D).
// ---------------------------------------------------------------------------
__global__ __launch_bounds__(256) void attn_core(
    const unsigned short* __restrict__ Qb16, const unsigned short* __restrict__ Kb16,
    const unsigned short* __restrict__ Vt16, const float* __restrict__ mask,
    unsigned short* __restrict__ AO) {
  extern __shared__ float smemf[];
  float* sc = smemf;                                  // [16][2048] fp32
  unsigned short* pb = (unsigned short*)smemf;        // bf16, row stride 4096

  const int tid  = threadIdx.x;
  const int lane = tid & 31;
  const int wave = tid >> 5;
  const int hi   = lane >> 4;
  const int l15  = lane & 15;

  const int bid = blockIdx.x;
  const int qt  = bid & 127;           // S/16 = 128 query tiles
  const int h   = (bid >> 7) & 15;
  const int b   = bid >> 11;
  const int qb  = qt * 16;
  const int kvh = h >> 1;              // N_REP = 2

  const unsigned short* Q  = Qb16 + (((size_t)(b * H_ + h)) * S_ + qb) * HD_;
  const unsigned short* Kp = Kb16 + ((size_t)(b * KVH_ + kvh)) * S_ * HD_;
  const unsigned short* Vt = Vt16 + ((size_t)(b * KVH_ + kvh)) * HD_ * S_;
  const float* Mrow = mask + ((size_t)b * S_ + qb) * S_;
  const float scale = 0.08838834764831845f;  // 1/sqrt(128)
  const v8f z8 = {0.f, 0.f, 0.f, 0.f, 0.f, 0.f, 0.f, 0.f};

  // ---- Phase 1: scores ----
  FragBF afr[4];
  #pragma unroll
  for (int ks = 0; ks < 4; ++ks) {
    int kb = ks * 32;
    const unsigned short* qp = Q + (size_t)l15 * HD_ + kb;
    afr[ks].q[0] = *(const uint4*)(qp + hi * 8);
    afr[ks].q[1] = *(const uint4*)(qp + 16 + hi * 8);
  }
  for (int ct = 0; ct < 16; ++ct) {
    int n0c = wave * 256 + ct * 16;
    v8f acc = z8;
    #pragma unroll
    for (int ks = 0; ks < 4; ++ks) {
      int kb = ks * 32;
      FragBF bf;
      const unsigned short* kp = Kp + (size_t)(n0c + l15) * HD_ + kb + hi * 8;
      bf.q[0] = ((const uint4*)kp)[0];
      bf.q[1] = ((const uint4*)kp)[2];
      acc = __builtin_amdgcn_wmma_f32_16x16x32_bf16(
          false, afr[ks].v, false, bf.v, (short)0, acc, false, false);
    }
    int col = n0c + l15;
    #pragma unroll
    for (int r = 0; r < 8; ++r) {
      int m = r + hi * 8;
      sc[m * 2048 + col] = acc[r] * scale + Mrow[(size_t)m * S_ + col];
    }
  }
  __syncthreads();

  // ---- Phase 2: exact top-k threshold + softmax (2 rows per wave) ----
  for (int rr = 0; rr < 2; ++rr) {
    int rowm = wave * 2 + rr;
    unsigned kv[64];
    #pragma unroll
    for (int i = 0; i < 64; ++i) kv[i] = fkey(sc[rowm * 2048 + lane + i * 32]);

    unsigned kmax = 0;
    #pragma unroll
    for (int i = 0; i < 64; ++i) kmax = (kv[i] > kmax) ? kv[i] : kmax;
    kmax = wred_max_u(kmax);
    float fm = fkey_inv(kmax);

    unsigned thr = 0;
    for (int bit = 31; bit >= 0; --bit) {
      unsigned cand = thr | (1u << bit);
      int cnt = 0;
      #pragma unroll
      for (int i = 0; i < 64; ++i) cnt += (kv[i] >= cand) ? 1 : 0;
      cnt = wred_sum_i(cnt);
      if (cnt >= TOPK_) thr = cand;
    }

    float pv[64];
    float sum = 0.f;
    #pragma unroll
    for (int i = 0; i < 64; ++i) {
      float e = 0.f;
      if (kv[i] >= thr) e = __expf(fkey_inv(kv[i]) - fm);
      pv[i] = e;
      sum += e;
    }
    sum = wred_sum_f(sum);
    float inv = 1.0f / sum;
    #pragma unroll
    for (int i = 0; i < 64; ++i)
      pb[rowm * 4096 + lane + i * 32] = f2bf(pv[i] * inv);
  }
  __syncthreads();

  // ---- Phase 3: out = P @ V (one 16x16 N-tile per wave, HD = 8*16) ----
  int n0 = wave * 16;
  v8f oacc = z8;
  for (int kb = 0; kb < S_; kb += 32) {
    FragBF a, bf;
    a.q[0] = *(const uint4*)&pb[l15 * 4096 + kb + hi * 8];
    a.q[1] = *(const uint4*)&pb[l15 * 4096 + kb + 16 + hi * 8];
    const unsigned short* vp = Vt + (size_t)(n0 + l15) * S_ + kb + hi * 8;
    bf.q[0] = ((const uint4*)vp)[0];
    bf.q[1] = ((const uint4*)vp)[2];
    oacc = __builtin_amdgcn_wmma_f32_16x16x32_bf16(
        false, a.v, false, bf.v, (short)0, oacc, false, false);
  }
  int colh = n0 + l15;
  #pragma unroll
  for (int r = 0; r < 8; ++r) {
    int m = r + hi * 8;
    AO[((size_t)(b * S_ + qb + m)) * D_ + h * HD_ + colh] = f2bf(oacc[r]);
  }
}

// ---------------------------------------------------------------------------
extern "C" void kernel_launch(void* const* d_in, const int* in_sizes, int n_in,
                              void* d_out, int out_size, void* d_ws, size_t ws_size,
                              hipStream_t stream) {
  (void)in_sizes; (void)n_in; (void)out_size; (void)ws_size;
  const float* hs   = (const float*)d_in[0];
  const float* mask = (const float*)d_in[1];
  const int*   pos  = (const int*)d_in[2];
  const float* q_w  = (const float*)d_in[3];
  const float* q_b  = (const float*)d_in[4];
  const float* k_w  = (const float*)d_in[5];
  const float* k_b  = (const float*)d_in[6];
  const float* v_w  = (const float*)d_in[7];
  const float* v_b  = (const float*)d_in[8];
  const float* o_w  = (const float*)d_in[9];
  const float* o_b  = (const float*)d_in[10];

  char* ws = (char*)d_ws;
  float* qf = (float*)ws;                                     // 33,554,432 B
  float* kf = (float*)(ws + 33554432);                        // 16,777,216 B
  float* vf = (float*)(ws + 50331648);                        // 16,777,216 B
  unsigned short* qb16 = (unsigned short*)(ws + 67108864);    // 16,777,216 B
  unsigned short* kb16 = (unsigned short*)(ws + 83886080);    //  8,388,608 B
  unsigned short* vt16 = (unsigned short*)(ws + 92274688);    //  8,388,608 B
  unsigned short* hsb  = (unsigned short*)(ws + 100663296);   // 16,777,216 B
  unsigned short* qwb  = (unsigned short*)(ws + 117440512);   //  8,388,608 B
  unsigned short* kwb  = (unsigned short*)(ws + 125829120);   //  4,194,304 B
  unsigned short* vwb  = (unsigned short*)(ws + 130023424);   //  4,194,304 B
  unsigned short* owb  = (unsigned short*)(ws + 134217728);   //  8,388,608 B
  unsigned short* aob  = (unsigned short*)ws;                 // overlays dead qf

  const int M = B_ * S_;  // 4096

  // One-time fp32 -> bf16 packing (keeps GEMM staging pure async b128 copies)
  to_bf16<<<(B_ * S_ * D_) / 1024, 256, 0, stream>>>(hs, hsb);
  to_bf16<<<(D_ * D_) / 1024, 256, 0, stream>>>(q_w, qwb);
  to_bf16<<<(KVH_ * HD_ * D_) / 1024, 256, 0, stream>>>(k_w, kwb);
  to_bf16<<<(KVH_ * HD_ * D_) / 1024, 256, 0, stream>>>(v_w, vwb);
  to_bf16<<<(D_ * D_) / 1024, 256, 0, stream>>>(o_w, owb);

  // Q/K/V projections (async double-buffered LDS staging, WMMA, fp32 out)
  gemm_bias<<<dim3(D_ / 128, M / 128), 256, 0, stream>>>(
      hsb, qwb, q_b, qf, M, D_, D_);
  gemm_bias<<<dim3((KVH_ * HD_) / 128, M / 128), 256, 0, stream>>>(
      hsb, kwb, k_b, kf, M, KVH_ * HD_, D_);
  gemm_bias<<<dim3((KVH_ * HD_) / 128, M / 128), 256, 0, stream>>>(
      hsb, vwb, v_b, vf, M, KVH_ * HD_, D_);

  // RoPE + bf16 packing (V transposed)
  rope_pack<<<65536, 256, 0, stream>>>(qf, kf, vf, pos, qb16, kb16, vt16);

  // Attention core: 128 KB dynamic LDS per workgroup (CDNA5 has 320 KB/WGP)
  attn_core<<<B_ * H_ * (S_ / 16), 256, 131072, stream>>>(
      qb16, kb16, vt16, mask, aob);

  // O-projection straight into d_out (fp32)
  gemm_bias<<<dim3(D_ / 128, M / 128), 256, 0, stream>>>(
      aob, owb, o_b, (float*)d_out, M, D_, D_);
}
